// CrossAttention_82343112999000
// MI455X (gfx1250) — matrix operand order repaired and checked by reading, not compiled, not run on hardware
//
#include <hip/hip_runtime.h>

typedef _Float16 f16;
typedef __attribute__((ext_vector_type(16))) _Float16 v16h;
typedef __attribute__((ext_vector_type(8)))  _Float16 v8h;
typedef __attribute__((ext_vector_type(8)))  float    v8f;

#define HEADS 16
#define HD    64
#define DIMSZ 1024
#define CTXSZ 768
#define NB    2
#define LQ    2048
#define LCC   1024

// ---------- WMMA helpers ----------
__device__ __forceinline__ v8f wmma32(v16h a, v16h b, v8f c) {
  // D = A(16x32 f16) * B(32x16 f16) + C(16x16 f32)
  return __builtin_amdgcn_wmma_f32_16x16x32_f16(false, a, false, b, (short)0, c, false, false);
}

// A-frag: lane(m = lane&15, half = lane>>4). p points at row_m + k0 + half*8.
// elements 0..7 = K[half*8 .. +8), elements 8..15 = K[16+half*8 .. +8)
__device__ __forceinline__ v16h load_afrag(const f16* p) {
  v8h lo = *(const v8h*)p;
  v8h hi = *(const v8h*)(p + 16);
  v16h r;
#pragma unroll
  for (int i = 0; i < 8; ++i) { r[i] = lo[i]; r[i + 8] = hi[i]; }
  return r;
}

// B-frag: lane(n = lane&15, half = lane>>4). p points at Wrow_n + k0 + half*8.
// elements 0..7 = K[half*8 .. +8), elements 8..15 = K[16+half*8 .. +8)
__device__ __forceinline__ v16h load_bfrag(const f16* p) {
  v8h lo = *(const v8h*)p;
  v8h hi = *(const v8h*)(p + 16);
  v16h r;
#pragma unroll
  for (int i = 0; i < 8; ++i) { r[i] = lo[i]; r[i + 8] = hi[i]; }
  return r;
}

// ---------- elementwise kernels ----------
__global__ void __launch_bounds__(256) cvt_f32_to_f16(const float* __restrict__ in,
                                                      f16* __restrict__ out, int n) {
  int i = blockIdx.x * blockDim.x + threadIdx.x;
  if (i < n) out[i] = (f16)in[i];
}

// V [NB*LCC][DIMSZ] f16  ->  Vt [NB][HEADS][HD][LCC] f16
__global__ void __launch_bounds__(256) transpose_v(const f16* __restrict__ V,
                                                   f16* __restrict__ Vt) {
  int idx = blockIdx.x * blockDim.x + threadIdx.x;   // 2^21 total
  int kk = idx & (LCC - 1);
  int d  = (idx >> 10) & (HD - 1);
  int h  = (idx >> 16) & (HEADS - 1);
  int b  = idx >> 20;
  Vt[idx] = V[(size_t)(b * LCC + kk) * DIMSZ + h * HD + d];
}

// ---------- generic GEMM: Y[M][N] = A[M][K] * W[N][K]^T ----------
// 4 waves / block; wave w owns rows [by*128 + w*32, +32), cols [bx*64, +64)
// 8 WMMA per k-step (2 A-frags x 4 B-frags), B-frags reused across both A rows.
template <int F32OUT>
__global__ void __launch_bounds__(128) gemm_xwt(const f16* __restrict__ A,
                                                const f16* __restrict__ W,
                                                f16* __restrict__ outH,
                                                float* __restrict__ outF,
                                                const float* __restrict__ bias,
                                                const float* __restrict__ resid,
                                                int M, int N, int K) {
  const int lane = threadIdx.x & 31;
  const int wave = threadIdx.x >> 5;
  const int m0   = blockIdx.y * 128 + wave * 32;
  const int n0   = blockIdx.x * 64;
  const int m    = lane & 15;
  const int half = lane >> 4;

  v8f acc[2][4];
#pragma unroll
  for (int u = 0; u < 2; ++u)
#pragma unroll
    for (int t = 0; t < 4; ++t) acc[u][t] = {};

  const f16* arow0 = A + (size_t)(m0 + m) * K;
  const f16* arow1 = A + (size_t)(m0 + 16 + m) * K;
  for (int k0 = 0; k0 < K; k0 += 32) {
    v16h a0 = load_afrag(arow0 + k0 + half * 8);
    v16h a1 = load_afrag(arow1 + k0 + half * 8);
#pragma unroll
    for (int t = 0; t < 4; ++t) {
      const f16* wp = W + (size_t)(n0 + t * 16 + m) * K + k0 + half * 8;
      v16h bf = load_bfrag(wp);
      acc[0][t] = wmma32(a0, bf, acc[0][t]);
      acc[1][t] = wmma32(a1, bf, acc[1][t]);
    }
  }

#pragma unroll
  for (int u = 0; u < 2; ++u) {
#pragma unroll
    for (int t = 0; t < 4; ++t) {
#pragma unroll
      for (int r = 0; r < 8; ++r) {
        int mm = m0 + u * 16 + half * 8 + r;
        int nn = n0 + t * 16 + m;
        if (F32OUT) {
          outF[(size_t)mm * N + nn] = acc[u][t][r] + bias[nn] + resid[(size_t)mm * N + nn];
        } else {
          outH[(size_t)mm * N + nn] = (f16)acc[u][t][r];
        }
      }
    }
  }
}

// ---------- attention: one wave per 16-query tile ----------
// Q  [NB*LQ][DIMSZ] f16, Kp [NB*LCC][DIMSZ] f16, Vt [NB][HEADS][HD][LCC] f16
// O  [NB*LQ][DIMSZ] f16
__global__ void __launch_bounds__(128) attn_kernel(const f16* __restrict__ Q,
                                                   const f16* __restrict__ Kp,
                                                   const f16* __restrict__ Vt,
                                                   f16* __restrict__ O) {
  extern __shared__ _Float16 sP[];           // 4 waves * 16 * LCC f16 = 128 KB
  const int lane = threadIdx.x & 31;
  const int wave = threadIdx.x >> 5;
  _Float16* P = sP + (size_t)wave * 16 * LCC;

  const int b    = blockIdx.z;
  const int h    = blockIdx.y;
  const int q0   = (blockIdx.x * 4 + wave) * 16;
  const int m    = lane & 15;
  const int half = lane >> 4;
  const float scale = 0.125f;                // HEAD_DIM^-0.5

  const f16* qbase = Q  + (size_t)(b * LQ)  * DIMSZ + h * HD;
  const f16* kbase = Kp + (size_t)(b * LCC) * DIMSZ + h * HD;
  const f16* vbase = Vt + ((size_t)(b * HEADS + h) * HD) * LCC;

  // Q A-frags for both k-steps of HEAD_DIM (64 = 2 * 32), kept in VGPRs
  v16h aq[2];
#pragma unroll
  for (int ks = 0; ks < 2; ++ks)
    aq[ks] = load_afrag(qbase + (size_t)(q0 + m) * DIMSZ + ks * 32 + half * 8);

  // ---- pass A: S = scale * Q K^T, stored f16 to per-wave LDS slice ----
  for (int n0 = 0; n0 < LCC; n0 += 16) {
    v8f s = {};
#pragma unroll
    for (int ks = 0; ks < 2; ++ks) {
      v16h bf = load_bfrag(kbase + (size_t)(n0 + m) * DIMSZ + ks * 32 + half * 8);
      s = wmma32(aq[ks], bf, s);
    }
#pragma unroll
    for (int r = 0; r < 8; ++r)
      P[(half * 8 + r) * LCC + n0 + m] = (_Float16)(s[r] * scale);
  }
  __builtin_amdgcn_wave_barrier();

  // ---- softmax: 2 lanes per row, 512 cols each, vectorized b128 LDS ops ----
  // Pass 1: row max. Pass 2: store exp(s-max) back (un-normalized, in (0,1]),
  // accumulate row sum. Normalization by 1/sum is deferred to the PV epilogue.
  const int row  = lane >> 1;
  const int part = lane & 1;
  _Float16* Sr = P + row * LCC + part * 512;
  float mx = -1e30f;
  for (int c = 0; c < 512; c += 8) {
    v8h v = *(const v8h*)(Sr + c);
#pragma unroll
    for (int i = 0; i < 8; ++i) mx = fmaxf(mx, (float)v[i]);
  }
  mx = fmaxf(mx, __shfl_xor(mx, 1));
  float sm = 0.f;
  for (int c = 0; c < 512; c += 8) {
    v8h v = *(const v8h*)(Sr + c);
    v8h e;
#pragma unroll
    for (int i = 0; i < 8; ++i) {
      float ev = __expf((float)v[i] - mx);
      sm += ev;
      e[i] = (_Float16)ev;
    }
    *(v8h*)(Sr + c) = e;
  }
  sm += __shfl_xor(sm, 1);
  float inv = 1.0f / sm;                     // lanes (2r, 2r+1) hold row r's 1/sum
  __builtin_amdgcn_wave_barrier();

  // ---- pass B: O = P * V  (A-frags from LDS, B-frags from Vt rows) ----
  v8f oacc[4];
#pragma unroll
  for (int t = 0; t < 4; ++t) oacc[t] = {};
  for (int k0 = 0; k0 < LCC; k0 += 32) {
    v16h af = load_afrag(P + m * LCC + k0 + half * 8);
#pragma unroll
    for (int t = 0; t < 4; ++t) {
      v16h bf = load_bfrag(vbase + (size_t)(t * 16 + m) * LCC + k0 + half * 8);
      oacc[t] = wmma32(af, bf, oacc[t]);
    }
  }

  f16* obase = O + (size_t)(b * LQ + q0) * DIMSZ + h * HD;
#pragma unroll
  for (int r = 0; r < 8; ++r) {
    float sc = __shfl(inv, (half * 8 + r) << 1);   // 1/rowsum for output row
#pragma unroll
    for (int t = 0; t < 4; ++t)
      obase[(size_t)(half * 8 + r) * DIMSZ + t * 16 + m] = (f16)(oacc[t][r] * sc);
  }
}

// ---------- host launch ----------
extern "C" void kernel_launch(void* const* d_in, const int* in_sizes, int n_in,
                              void* d_out, int out_size, void* d_ws, size_t ws_size,
                              hipStream_t stream) {
  const float* x   = (const float*)d_in[0];   // [2,2048,1024]
  const float* ctx = (const float*)d_in[1];   // [2,1024,768]
  const float* Wq  = (const float*)d_in[2];   // [1024,1024]
  const float* Wk  = (const float*)d_in[3];   // [1024,768]
  const float* Wv  = (const float*)d_in[4];   // [1024,768]
  const float* Wo  = (const float*)d_in[5];   // [1024,1024]
  const float* bo  = (const float*)d_in[6];   // [1024]
  float* out = (float*)d_out;

  const int nX  = NB * LQ * DIMSZ;       // 4194304
  const int nC  = NB * LCC * CTXSZ;      // 1572864
  const int nWq = DIMSZ * DIMSZ;         // 1048576
  const int nWk = DIMSZ * CTXSZ;         // 786432
  const int nKV = NB * LCC * DIMSZ;      // 2097152

  size_t off = 0;
  auto alloc = [&](size_t bytes) { size_t o = off; off = (off + bytes + 255) & ~(size_t)255; return o; };
  char* ws = (char*)d_ws;
  f16* xh  = (f16*)(ws + alloc((size_t)nX * 2));
  f16* ch  = (f16*)(ws + alloc((size_t)nC * 2));
  f16* wqh = (f16*)(ws + alloc((size_t)nWq * 2));
  f16* wkh = (f16*)(ws + alloc((size_t)nWk * 2));
  f16* wvh = (f16*)(ws + alloc((size_t)nWk * 2));
  f16* woh = (f16*)(ws + alloc((size_t)nWq * 2));
  f16* Qh  = (f16*)(ws + alloc((size_t)nX * 2));
  f16* Kh  = (f16*)(ws + alloc((size_t)nKV * 2));
  f16* Vh  = (f16*)(ws + alloc((size_t)nKV * 2));
  f16* Vth = (f16*)(ws + alloc((size_t)nKV * 2));
  f16* Oh  = (f16*)(ws + alloc((size_t)nX * 2));

  // 1) convert to f16
  cvt_f32_to_f16<<<(nX  + 255) / 256, 256, 0, stream>>>(x,   xh,  nX);
  cvt_f32_to_f16<<<(nC  + 255) / 256, 256, 0, stream>>>(ctx, ch,  nC);
  cvt_f32_to_f16<<<(nWq + 255) / 256, 256, 0, stream>>>(Wq,  wqh, nWq);
  cvt_f32_to_f16<<<(nWk + 255) / 256, 256, 0, stream>>>(Wk,  wkh, nWk);
  cvt_f32_to_f16<<<(nWk + 255) / 256, 256, 0, stream>>>(Wv,  wvh, nWk);
  cvt_f32_to_f16<<<(nWq + 255) / 256, 256, 0, stream>>>(Wo,  woh, nWq);

  // 2) projections: Q = x Wq^T ; K = ctx Wk^T ; V = ctx Wv^T
  gemm_xwt<0><<<dim3(DIMSZ / 64, (NB * LQ) / 128), 128, 0, stream>>>(
      xh, wqh, Qh, nullptr, nullptr, nullptr, NB * LQ, DIMSZ, DIMSZ);
  gemm_xwt<0><<<dim3(DIMSZ / 64, (NB * LCC) / 128), 128, 0, stream>>>(
      ch, wkh, Kh, nullptr, nullptr, nullptr, NB * LCC, DIMSZ, CTXSZ);
  gemm_xwt<0><<<dim3(DIMSZ / 64, (NB * LCC) / 128), 128, 0, stream>>>(
      ch, wvh, Vh, nullptr, nullptr, nullptr, NB * LCC, DIMSZ, CTXSZ);

  // 3) V -> Vt [b][h][d][kk]
  transpose_v<<<(nKV + 255) / 256, 256, 0, stream>>>(Vh, Vth);

  // 4) attention (dynamic LDS: 4 waves * 16 * 1024 f16 = 128 KB)
  attn_kernel<<<dim3(LQ / 64, HEADS, NB), 128, 4 * 16 * LCC * sizeof(f16), stream>>>(
      Qh, Kh, Vth, Oh);

  // 5) output projection + bias + residual (fp32 out)
  gemm_xwt<1><<<dim3(DIMSZ / 64, (NB * LQ) / 128), 128, 0, stream>>>(
      Oh, woh, nullptr, out, bo, x, NB * LQ, DIMSZ, DIMSZ);
}
